// HRLEncoder_38955353375315
// MI455X (gfx1250) — matrix operand order repaired and checked by reading, not compiled, not run on hardware
//
#include <hip/hip_runtime.h>
#include <hip/hip_bf16.h>

// ---------------------------------------------------------------------------
// Problem constants (match reference)
// ---------------------------------------------------------------------------
enum : int {
  cN0 = 400000, cN1 = 100000, cN2 = 20000, cN3 = 4000,
  cE1 = 400000, cE2 = 100000, cE3 = 20000,
  cD = 128, cG = 2048, cOUT = 256
};

typedef __attribute__((ext_vector_type(8)))  float  v8f;
typedef __attribute__((ext_vector_type(8)))  __bf16 v8bf;
typedef __attribute__((ext_vector_type(16))) __bf16 v16bf;

__device__ __forceinline__ unsigned short f2bf(float f) {
  unsigned int u = __float_as_uint(f);
  u += 0x7FFFu + ((u >> 16) & 1u);           // round-to-nearest-even
  return (unsigned short)(u >> 16);
}

__device__ __forceinline__ void atomic_add_f32(float* p, float v) {
  // atomic-without-return, agent scope -> global_atomic_add_f32 (STOREcnt path)
  __hip_atomic_fetch_add(p, v, __ATOMIC_RELAXED, __HIP_MEMORY_SCOPE_AGENT);
}

// ---------------------------------------------------------------------------
// Generic WMMA GEMM:  C[M,N] = A[M,K] @ B[K,N] + bias[N]   (optional ReLU)
// Block: 256 threads (8 waves). Block tile: 128 rows x 128 cols.
// K staged through LDS in 128-chunks, converted f32 -> bf16.
// Inner loop preloads all 8 B fragments per k-step, then issues 8 WMMAs
// back-to-back so a single s_wait_dscnt covers the whole batch.
// ---------------------------------------------------------------------------
#define LDSP 136   // padded LDS row stride in shorts (breaks bank alignment)

__global__ __launch_bounds__(256)
void gemm_wmma_kernel(const float* __restrict__ A, int M, int K,
                      const float* __restrict__ B, int N,
                      const float* __restrict__ bias,
                      float* __restrict__ C, int relu)
{
  __shared__ unsigned short As[128 * LDSP];   // [row][k]   bf16 bits
  __shared__ unsigned short Bs[128 * LDSP];   // [n][k]     bf16 bits (transposed)

  const int tid  = threadIdx.x;
  const int wave = tid >> 5;
  const int lane = tid & 31;
  const int half = lane >> 4;
  const int l16  = lane & 15;

  const int m0 = blockIdx.x * 128;
  const int n0 = blockIdx.y * 128;

  v8f acc[8];
#pragma unroll
  for (int t = 0; t < 8; ++t)
#pragma unroll
    for (int r = 0; r < 8; ++r) acc[t][r] = 0.0f;

  for (int k0 = 0; k0 < K; k0 += 128) {
    // ---- stage A (128x128) and B (128x128, transposed) into LDS as bf16 ----
#pragma unroll
    for (int i = 0; i < 16; ++i) {
      const int idx = tid + 256 * i;     // 0..4095 float4 slots
      const int row = idx >> 5;          // 0..127
      const int c4  = idx & 31;          // float4 index within row
      // A chunk
      const int grow = m0 + row;
      float4 av = make_float4(0.f, 0.f, 0.f, 0.f);
      if (grow < M)
        av = *reinterpret_cast<const float4*>(A + (size_t)grow * K + k0 + c4 * 4);
      *reinterpret_cast<ushort4*>(&As[row * LDSP + c4 * 4]) =
          make_ushort4(f2bf(av.x), f2bf(av.y), f2bf(av.z), f2bf(av.w));
      // B chunk (rows k0+row, cols n0 + c4*4) -> Bs[n][k]
      const float4 bv =
          *reinterpret_cast<const float4*>(B + (size_t)(k0 + row) * N + n0 + c4 * 4);
      Bs[(c4 * 4 + 0) * LDSP + row] = f2bf(bv.x);
      Bs[(c4 * 4 + 1) * LDSP + row] = f2bf(bv.y);
      Bs[(c4 * 4 + 2) * LDSP + row] = f2bf(bv.z);
      Bs[(c4 * 4 + 3) * LDSP + row] = f2bf(bv.w);
    }
    __syncthreads();

    // ---- compute: 4 k-steps of 32, 8 column tiles per wave ----
    const int arow = wave * 16 + l16;   // A fragment: M = lane%16 within wave strip
#pragma unroll
    for (int ks = 0; ks < 4; ++ks) {
      const int kb = ks * 32;
      // A 16-bit fragment: j<8 -> K=kb+half*8+j ; j>=8 -> K=kb+16+half*8+(j-8)
      const unsigned short* ap = &As[arow * LDSP + kb + half * 8];
      const v8bf alo = *reinterpret_cast<const v8bf*>(ap);
      const v8bf ahi = *reinterpret_cast<const v8bf*>(ap + 16);
      const v16bf afrag = __builtin_shufflevector(alo, ahi,
          0,1,2,3,4,5,6,7,8,9,10,11,12,13,14,15);
      // Preload all 8 B fragments for this k-step (independent ds_loads),
      // then issue the 8 WMMAs back-to-back.
      v16bf bfrag[8];
#pragma unroll
      for (int t = 0; t < 8; ++t) {
        // B fragment: N = lane%16, j<8 -> K=kb+half*8+j ; j>=8 -> K=kb+16+half*8+(j-8)
        const unsigned short* bp = &Bs[(t * 16 + l16) * LDSP + kb + half * 8];
        const v8bf blo = *reinterpret_cast<const v8bf*>(bp);
        const v8bf bhi = *reinterpret_cast<const v8bf*>(bp + 16);
        bfrag[t] = __builtin_shufflevector(blo, bhi,
            0,1,2,3,4,5,6,7,8,9,10,11,12,13,14,15);
      }
#pragma unroll
      for (int t = 0; t < 8; ++t) {
        acc[t] = __builtin_amdgcn_wmma_f32_16x16x32_bf16(
            false, afrag, false, bfrag[t], (short)0, acc[t], false, false);
      }
    }
    __syncthreads();
  }

  // ---- epilogue: D layout VGPR r -> M = half*8 + r, N = lane%16 ----
  const int mrow0 = m0 + wave * 16 + half * 8;
#pragma unroll
  for (int t = 0; t < 8; ++t) {
    const int col = n0 + t * 16 + l16;
    const float bv = bias[col];
#pragma unroll
    for (int r = 0; r < 8; ++r) {
      const int row = mrow0 + r;
      if (row < M) {
        float v = acc[t][r] + bv;
        if (relu) v = fmaxf(v, 0.0f);
        C[(size_t)row * N + col] = v;
      }
    }
  }
}

// ---------------------------------------------------------------------------
// Scatter-add: out[dst[i]][colOff + :D] += x[src[i]][:D]
// One wave per item, float4 per lane (32 lanes * 4 = 128 floats).
// src == nullptr means identity (used for global_add_pool).
// ---------------------------------------------------------------------------
__global__ __launch_bounds__(256)
void scatter_add_kernel(const float* __restrict__ x,
                        const int* __restrict__ src,
                        const int* __restrict__ dst,
                        int count, float* __restrict__ out,
                        int ldOut, int colOff)
{
  const int item = (int)((blockIdx.x * blockDim.x + threadIdx.x) >> 5);
  const int lane = threadIdx.x & 31;
  if (item >= count) return;
  const int s = src ? src[item] : item;
  const int d = dst[item];
  const float4 v = reinterpret_cast<const float4*>(x + (size_t)s * cD)[lane];
  float* o = out + (size_t)d * ldOut + colOff + lane * 4;
  atomic_add_f32(o + 0, v.x);
  atomic_add_f32(o + 1, v.y);
  atomic_add_f32(o + 2, v.z);
  atomic_add_f32(o + 3, v.w);
}

// ---------------------------------------------------------------------------
// Column statistics over z[M,128]: stats[0..127]=sum, stats[128..255]=sumsq
// ---------------------------------------------------------------------------
__global__ __launch_bounds__(256)
void colstats_kernel(const float* __restrict__ z, int M, float* __restrict__ stats)
{
  __shared__ float ssum[2][128];
  __shared__ float ssq[2][128];
  const int tid = threadIdx.x;
  const int c = tid & 127;
  const int g = tid >> 7;
  const int row0 = blockIdx.x * 256;
  float s = 0.f, q = 0.f;
  for (int r = g; r < 256; r += 2) {
    const int row = row0 + r;
    if (row < M) {
      const float v = z[(size_t)row * cD + c];
      s += v; q += v * v;
    }
  }
  ssum[g][c] = s; ssq[g][c] = q;
  __syncthreads();
  if (g == 0) {
    atomic_add_f32(&stats[c],       ssum[0][c] + ssum[1][c]);
    atomic_add_f32(&stats[128 + c], ssq[0][c]  + ssq[1][c]);
  }
}

// scale/shift from stats:  ss[0..127]=scale, ss[128..255]=shift
__global__ void bnfinalize_kernel(const float* __restrict__ stats, int M,
                                  const float* __restrict__ gamma,
                                  const float* __restrict__ beta,
                                  float* __restrict__ ss)
{
  const int c = threadIdx.x;
  if (c < 128) {
    const float invM = 1.0f / (float)M;
    const float mean = stats[c] * invM;
    const float var  = stats[128 + c] * invM - mean * mean;
    const float sc   = gamma[c] * rsqrtf(var + 1e-5f);
    ss[c]       = sc;
    ss[128 + c] = beta[c] - mean * sc;
  }
}

// y = relu(z * scale[c] + shift[c]), float4-vectorized (n multiple of 128)
__global__ __launch_bounds__(256)
void bnrelu_kernel(const float4* __restrict__ z4, const float* __restrict__ ss,
                   float4* __restrict__ y4, long long n4)
{
  const long long i = (long long)blockIdx.x * blockDim.x + threadIdx.x;
  if (i < n4) {
    const int c0 = (int)((i & 31) * 4);   // column of first element in this float4
    const float4 v = z4[i];
    float4 r;
    r.x = fmaxf(v.x * ss[c0 + 0] + ss[128 + c0 + 0], 0.0f);
    r.y = fmaxf(v.y * ss[c0 + 1] + ss[128 + c0 + 1], 0.0f);
    r.z = fmaxf(v.z * ss[c0 + 2] + ss[128 + c0 + 2], 0.0f);
    r.w = fmaxf(v.w * ss[c0 + 3] + ss[128 + c0 + 3], 0.0f);
    y4[i] = r;
  }
}

__global__ __launch_bounds__(256)
void zero4_kernel(float4* __restrict__ p, long long n4)
{
  long long i = (long long)blockIdx.x * blockDim.x + threadIdx.x;
  const long long stride = (long long)gridDim.x * blockDim.x;
  const float4 z = make_float4(0.f, 0.f, 0.f, 0.f);
  for (; i < n4; i += stride) p[i] = z;
}

// ---------------------------------------------------------------------------
// Host launcher
// ---------------------------------------------------------------------------
extern "C" void kernel_launch(void* const* d_in, const int* in_sizes, int n_in,
                              void* d_out, int out_size, void* d_ws, size_t ws_size,
                              hipStream_t stream)
{
  const float* x      = (const float*)d_in[0];
  const int*   src1   = (const int*)d_in[1];
  const int*   dst1   = (const int*)d_in[2];
  const int*   src2   = (const int*)d_in[3];
  const int*   dst2   = (const int*)d_in[4];
  const int*   src3   = (const int*)d_in[5];
  const int*   dst3   = (const int*)d_in[6];
  const int*   batch0 = (const int*)d_in[7];
  const int*   batch1 = (const int*)d_in[8];
  const int*   batch2 = (const int*)d_in[9];
  const int*   batch3 = (const int*)d_in[10];
  const float* W1a = (const float*)d_in[11];
  const float* b1a = (const float*)d_in[12];
  const float* g1a = (const float*)d_in[13];
  const float* be1a= (const float*)d_in[14];
  const float* W1b = (const float*)d_in[15];
  const float* b1b = (const float*)d_in[16];
  const float* g1b = (const float*)d_in[17];
  const float* be1b= (const float*)d_in[18];
  const float* W2a = (const float*)d_in[19];
  const float* b2a = (const float*)d_in[20];
  const float* g2a = (const float*)d_in[21];
  const float* be2a= (const float*)d_in[22];
  const float* W2b = (const float*)d_in[23];
  const float* b2b = (const float*)d_in[24];
  const float* g2b = (const float*)d_in[25];
  const float* be2b= (const float*)d_in[26];
  const float* W3a = (const float*)d_in[27];
  const float* b3a = (const float*)d_in[28];
  const float* W3b = (const float*)d_in[29];
  const float* b3b = (const float*)d_in[30];
  const float* Wout= (const float*)d_in[31];
  const float* bout= (const float*)d_in[32];

  // ---- workspace layout ----
  size_t off = 0;
  auto wsAlloc = [&](size_t bytes) -> float* {
    float* p = (float*)((char*)d_ws + off);
    off += (bytes + 255) & ~(size_t)255;
    return p;
  };
  float* P0     = wsAlloc((size_t)cN1 * cD * 4);   // ping
  float* P1     = wsAlloc((size_t)cN1 * cD * 4);   // pong
  float* X1     = wsAlloc((size_t)cN1 * cD * 4);
  float* X2     = wsAlloc((size_t)cN2 * cD * 4);
  float* X3     = wsAlloc((size_t)cN3 * cD * 4);
  float* pooled = wsAlloc((size_t)cG * 4 * cD * 4);
  float* stats  = wsAlloc(256 * 4);
  float* ss     = wsAlloc(256 * 4);
  (void)ws_size; (void)n_in; (void)in_sizes; (void)out_size;

  auto zero = [&](float* p, long long n) {   // n multiple of 4
    zero4_kernel<<<2048, 256, 0, stream>>>((float4*)p, n / 4);
  };
  auto scatter = [&](const float* xin, const int* s, const int* d, int count,
                     float* outp, int ldo, int coff) {
    scatter_add_kernel<<<(count + 7) / 8, 256, 0, stream>>>(xin, s, d, count, outp, ldo, coff);
  };
  auto gemm = [&](const float* A, int M, int K, const float* B, int N,
                  const float* bias, float* C, int relu) {
    dim3 grid((M + 127) / 128, N / 128);
    gemm_wmma_kernel<<<grid, 256, 0, stream>>>(A, M, K, B, N, bias, C, relu);
  };
  auto bn = [&](const float* z, int M, const float* gamma, const float* beta,
                float* y) {
    zero(stats, 256);
    colstats_kernel<<<(M + 255) / 256, 256, 0, stream>>>(z, M, stats);
    bnfinalize_kernel<<<1, 128, 0, stream>>>(stats, M, gamma, beta, ss);
    const long long n4 = (long long)M * cD / 4;
    bnrelu_kernel<<<(unsigned)((n4 + 255) / 256), 256, 0, stream>>>(
        (const float4*)z, ss, (float4*)y, n4);
  };

  // ---- layer 1 ----
  zero(P0, (long long)cN1 * cD);
  scatter(x, src1, dst1, cE1, P0, cD, 0);
  gemm(P0, cN1, cD, W1a, cD, b1a, P1, 0);
  bn(P1, cN1, g1a, be1a, P0);
  gemm(P0, cN1, cD, W1b, cD, b1b, P1, 0);
  bn(P1, cN1, g1b, be1b, X1);

  // ---- layer 2 ----
  zero(P0, (long long)cN2 * cD);
  scatter(X1, src2, dst2, cE2, P0, cD, 0);
  gemm(P0, cN2, cD, W2a, cD, b2a, P1, 0);
  bn(P1, cN2, g2a, be2a, P0);
  gemm(P0, cN2, cD, W2b, cD, b2b, P1, 0);
  bn(P1, cN2, g2b, be2b, X2);

  // ---- layer 3 (no BN, fused ReLU in GEMM) ----
  zero(P0, (long long)cN3 * cD);
  scatter(X2, src3, dst3, cE3, P0, cD, 0);
  gemm(P0, cN3, cD, W3a, cD, b3a, P1, 1);
  gemm(P1, cN3, cD, W3b, cD, b3b, X3, 1);

  // ---- pooling into [G, 4*D] ----
  zero(pooled, (long long)cG * 4 * cD);
  scatter(x,  nullptr, batch0, cN0, pooled, 4 * cD, 0);
  scatter(X1, nullptr, batch1, cN1, pooled, 4 * cD, cD);
  scatter(X2, nullptr, batch2, cN2, pooled, 4 * cD, 2 * cD);
  scatter(X3, nullptr, batch3, cN3, pooled, 4 * cD, 3 * cD);

  // ---- output head: [2048, 512] @ [512, 256] + bout ----
  gemm(pooled, cG, 4 * cD, Wout, cOUT, bout, (float*)d_out, 0);
}
